// RQLayer_57286273794525
// MI455X (gfx1250) — matrix operand order repaired and checked by reading, not compiled, not run on hardware
//
#include <hip/hip_runtime.h>
#include <hip/hip_bf16.h>
#include <cfloat>

// ---------------------------------------------------------------------------
// Residual VQ (RQLayer forward), MI455X / gfx1250, wave32 + bf16 WMMA.
//   dist ranking:  score = 0.5*|c|^2 - x.c   (x.c via v_wmma_f32_16x16x32_bf16)
//   outputs (quantized, loss, residual) computed in f32 from f32 codebooks.
//   v2: 2-deep B-fragment register pipeline + dual independent WMMA
//       accumulator chains to remove s_wait_dscnt 0x0 stalls before each WMMA.
// ---------------------------------------------------------------------------

typedef __attribute__((ext_vector_type(16))) __bf16 v16bf;
typedef __attribute__((ext_vector_type(8)))  float  v8f;

#define Bq      32768
#define Dq      256
#define Kq      2048
#define Lq      4
#define MT      128                 // rows per block
#define NBLK    (Bq / MT)           // 256 blocks
#define THREADS 256                 // 8 wave32
#define CBS     272                 // padded LDS row stride (bf16 elems, 544B)

#define LOSS_SCALE (0.25f / (32768.0f * 256.0f * 4.0f))   // BETA/(B*D*L)

// ---------------------------------------------------------------------------
// Prep: f32 codebooks -> bf16 in ws, 0.5*|c|^2 per code, zero count histogram.
// grid = L*K blocks of 256 threads (one block per codebook row).
// ---------------------------------------------------------------------------
__global__ __launch_bounds__(256) void vq_prep(const float* __restrict__ cb,
                                               __bf16* __restrict__ cb_bf,
                                               float* __restrict__ cn2h,
                                               int* __restrict__ counts) {
  __shared__ float red[256];
  const int row = blockIdx.x;               // 0 .. L*K-1
  const int t   = threadIdx.x;
  float v = cb[(size_t)row * Dq + t];
  cb_bf[(size_t)row * Dq + t] = (__bf16)v;
  red[t] = v * v;
  __syncthreads();
  for (int s = 128; s > 0; s >>= 1) {
    if (t < s) red[t] += red[t + s];
    __syncthreads();
  }
  if (t == 0) { cn2h[row] = 0.5f * red[0]; counts[row] = 0; }
}

// ---------------------------------------------------------------------------
// One residual level: distance GEMM (bf16 WMMA) + argmin + gather + update.
// ---------------------------------------------------------------------------
__global__ __launch_bounds__(256) void vq_level(
    const float*  __restrict__ x,      const float* __restrict__ cb,
    const __bf16* __restrict__ cb_bf,  const float* __restrict__ cn2h,
    int*   __restrict__ counts,        float* __restrict__ lossPartial,
    float* __restrict__ qout,          int*   __restrict__ outInds,
    int level) {
  __shared__ __align__(64) unsigned char smem[64 * 1024];
  __bf16* xbf    = (__bf16*)smem;                       // phase 1: 128*256 bf16 = 64KB
  __bf16* cbuf0  = (__bf16*)smem;                       // phase 2: 16*CBS*2 = 8704B
  __bf16* cbuf1  = (__bf16*)(smem + 16 * CBS * 2);      //          8704B
  int*    rowidx = (int*)  (smem + 20 * 1024);          // 128 ints
  float*  red    = (float*)(smem + 24 * 1024);          // 256 floats

  const int tid  = threadIdx.x;
  const int lane = tid & 31;
  const int wave = tid >> 5;
  const int hf   = lane >> 4;                 // which 16-lane half
  const int row0 = blockIdx.x * MT;
  const int lK   = level * Kq;

  // ---- phase 1: stage residual r = x - q_acc as bf16 (coalesced) ----------
  for (int i = tid; i < MT * Dq; i += THREADS) {
    size_t g = (size_t)row0 * Dq + i;
    float  q = (level == 0) ? 0.0f : qout[g];
    xbf[i] = (__bf16)(x[g] - q);
  }
  __syncthreads();

  // ---- preload A fragments (contiguous k order, same as B) ------------
  // lanes<16: elems 0..15 -> K 0..15
  // lanes>=16: elems 0..15 -> K 16..31
  const int rloc = wave * 16 + (lane & 15);   // this lane's A row (local)
  const int aoff = hf ? 16 : 0;
  v16bf a_frag[8];
#pragma unroll
  for (int kd = 0; kd < 8; ++kd) {
    union { v16bf v; uint4 u[2]; } tmp;
    const __bf16* p = xbf + rloc * Dq + kd * 32;
    tmp.u[0] = *(const uint4*)(p + aoff);
    tmp.u[1] = *(const uint4*)(p + 8 + aoff);
    a_frag[kd] = tmp.v;
  }
  __syncthreads();                            // xbf region dead beyond here

  float rVal[8];
  int   rIdx[8];
#pragma unroll
  for (int j = 0; j < 8; ++j) { rVal[j] = FLT_MAX; rIdx[j] = 0; }

  const int colloc = lane & 15;               // this lane's N slot
  const int boff   = hf ? 16 : 0;             // B: lanes<16 K 0..15, >=16 K 16..31

  // ---- prologue: load codebook chunk 0 into cbuf0 (coalesced, padded) -----
  {
    const __bf16* src = cb_bf + (size_t)lK * Dq;
    for (int i = tid; i < 512; i += THREADS) {            // 512 x uint4 = 8KB
      int r = i >> 5, c = i & 31;
      *(uint4*)(cbuf0 + r * CBS + c * 8) = *(const uint4*)(src + r * Dq + c * 8);
    }
  }
  __syncthreads();

  // ---- main loop: 128 chunks of 16 codes, double-buffered -----------------
  for (int nb = 0; nb < Kq / 16; ++nb) {
    __bf16* cur = (nb & 1) ? cbuf1 : cbuf0;
    __bf16* nxt = (nb & 1) ? cbuf0 : cbuf1;
    if (nb + 1 < Kq / 16) {
      const __bf16* src = cb_bf + ((size_t)lK + (size_t)(nb + 1) * 16) * Dq;
      for (int i = tid; i < 512; i += THREADS) {
        int r = i >> 5, c = i & 31;
        *(uint4*)(nxt + r * CBS + c * 8) = *(const uint4*)(src + r * Dq + c * 8);
      }
    }

    // dot tile: 8 x v_wmma_f32_16x16x32_bf16 over D=256.
    //  - two independent accumulator chains (even/odd kd) so consecutive
    //    WMMAs have no C-operand RAW dependency
    //  - B fragments prefetched one pipeline stage (2 frags) ahead so the
    //    consuming WMMA never waits on dscnt==0
    const __bf16* bp = cur + colloc * CBS + boff;
    v8f acc0 = {0.f, 0.f, 0.f, 0.f, 0.f, 0.f, 0.f, 0.f};
    v8f acc1 = acc0;
    v16bf b0 = *(const v16bf*)(bp + 0 * 32);
    v16bf b1 = *(const v16bf*)(bp + 1 * 32);
#pragma unroll
    for (int kd = 0; kd < 8; kd += 2) {
      v16bf bn0 = b0, bn1 = b1;
      if (kd + 2 < 8) {
        bn0 = *(const v16bf*)(bp + (kd + 2) * 32);
        bn1 = *(const v16bf*)(bp + (kd + 3) * 32);
      }
      acc0 = __builtin_amdgcn_wmma_f32_16x16x32_bf16(
          false, a_frag[kd],     false, b0, (short)0, acc0, false, false);
      acc1 = __builtin_amdgcn_wmma_f32_16x16x32_bf16(
          false, a_frag[kd + 1], false, b1, (short)0, acc1, false, false);
      b0 = bn0;
      b1 = bn1;
    }

    // running argmin: dot(row wave*16 + hf*8 + j, col nb*16+colloc)
    const int gcol = nb * 16 + colloc;
    const float sc = cn2h[lK + gcol];
#pragma unroll
    for (int j = 0; j < 8; ++j) {
      float s = sc - (acc0[j] + acc1[j]);
      if (s < rVal[j]) { rVal[j] = s; rIdx[j] = gcol; }   // strict '<' => first-min
    }
    __syncthreads();                                      // nxt ready / cur reusable
  }

  // ---- combine the 16 column slots per row (butterfly in each half) -------
#pragma unroll
  for (int j = 0; j < 8; ++j) {
    float v = rVal[j];
    int   id = rIdx[j];
#pragma unroll
    for (int m = 1; m < 16; m <<= 1) {
      float ov = __shfl_xor(v, m, 32);
      int   oi = __shfl_xor(id, m, 32);
      if (ov < v || (ov == v && oi < id)) { v = ov; id = oi; }
    }
    if ((lane & 15) == 0) rowidx[wave * 16 + hf * 8 + j] = id;
  }
  __syncthreads();

  // ---- epilogue: gather f32 codeword, update quantized, loss, indices -----
  float lsum = 0.f;
  for (int i = tid; i < MT * Dq; i += THREADS) {
    int   r    = i >> 8;
    int   ind  = rowidx[r];
    size_t g   = (size_t)row0 * Dq + i;
    float cv   = cb[((size_t)lK + (size_t)ind) * Dq + (i & 255)];
    float qprev = (level == 0) ? 0.0f : qout[g];
    float res  = x[g] - qprev;                // residual at this level
    qout[g]    = qprev + cv;                  // quantized += x_q
    float df   = res - cv;
    lsum += df * df;
  }
  if (tid < MT) {
    int ind = rowidx[tid];
    outInds[(size_t)(row0 + tid) * Lq + level] = ind;     // inds[b][l]
    atomicAdd(&counts[lK + ind], 1);
  }
  red[tid] = lsum;
  __syncthreads();
  for (int s = 128; s > 0; s >>= 1) {
    if (tid < s) red[tid] += red[tid + s];
    __syncthreads();
  }
  if (tid == 0) {
    float c = red[0] * LOSS_SCALE;
    if (level == 0) lossPartial[blockIdx.x] = c;          // deterministic accum
    else            lossPartial[blockIdx.x] += c;
  }
}

// ---------------------------------------------------------------------------
// Finalize: fixed-order loss reduction + unused-code count.
// ---------------------------------------------------------------------------
__global__ __launch_bounds__(256) void vq_finalize(
    const int* __restrict__ counts, const float* __restrict__ lossPartial,
    float* __restrict__ lossOut, int* __restrict__ unusedOut) {
  __shared__ float fr[256];
  __shared__ int   ir[256];
  const int t = threadIdx.x;
  int uz = 0;
  for (int i = t; i < Lq * Kq; i += 256) uz += (counts[i] == 0) ? 1 : 0;
  fr[t] = lossPartial[t];                     // NBLK == 256
  ir[t] = uz;
  __syncthreads();
  for (int s = 128; s > 0; s >>= 1) {
    if (t < s) { fr[t] += fr[t + s]; ir[t] += ir[t + s]; }
    __syncthreads();
  }
  if (t == 0) { lossOut[0] = fr[0]; unusedOut[0] = ir[0]; }
}

// ---------------------------------------------------------------------------
extern "C" void kernel_launch(void* const* d_in, const int* in_sizes, int n_in,
                              void* d_out, int out_size, void* d_ws, size_t ws_size,
                              hipStream_t stream) {
  const float* x  = (const float*)d_in[0];    // [B, D] f32
  const float* cb = (const float*)d_in[1];    // [L, K, D] f32

  // workspace layout
  char*   ws          = (char*)d_ws;
  __bf16* cb_bf       = (__bf16*)ws;                                  // 4 MB
  float*  cn2h        = (float*)(ws + (size_t)Lq * Kq * Dq * 2);      // 32 KB
  int*    counts      = (int*)((char*)cn2h + (size_t)Lq * Kq * 4);    // 32 KB
  float*  lossPartial = (float*)((char*)counts + (size_t)Lq * Kq * 4);// 1 KB

  // output layout: [quantized B*D f32][loss f32][unused i32][inds B*L i32]
  float* qout      = (float*)d_out;
  float* lossOut   = qout + (size_t)Bq * Dq;
  int*   unusedOut = (int*)(lossOut + 1);
  int*   outInds   = (int*)(lossOut + 2);

  vq_prep<<<Lq * Kq, 256, 0, stream>>>(cb, cb_bf, cn2h, counts);
  for (int l = 0; l < Lq; ++l) {
    vq_level<<<NBLK, THREADS, 0, stream>>>(x, cb, cb_bf, cn2h, counts,
                                           lossPartial, qout, outInds, l);
  }
  vq_finalize<<<1, 256, 0, stream>>>(counts, lossPartial, lossOut, unusedOut);
}
